// ConvCapsuleLayer_81406810128986
// MI455X (gfx1250) — matrix operand order repaired and checked by reading, not compiled, not run on hardware
//
#include <hip/hip_runtime.h>
#include <hip/hip_bf16.h>

// ---------------------------------------------------------------------------
// ConvCapsuleLayer for MI455X (gfx1250, wave32, WMMA)
//   conv (implicit GEMM, bf16 WMMA, f32 accum, 4 oc-tiles/wave) -> votes
//   dynamic routing (async global->LDS staging, wave32 shuffle reductions)
// ---------------------------------------------------------------------------

typedef __attribute__((ext_vector_type(16))) __bf16 v16bf;
typedef __attribute__((ext_vector_type(8)))  __bf16 v8bf;
typedef __attribute__((ext_vector_type(8)))  float  v8f;

#define BATCH     8
#define IN_CAPS   8
#define IN_ATOMS  32
#define NUM_CAPS  8
#define NUM_ATOMS 32
#define HSZ       64
#define WSZ       64
#define BC        (BATCH * IN_CAPS)      // 64
#define OC        (NUM_CAPS * NUM_ATOMS) // 256
#define KPOS      25                     // 5x5 kernel positions
#define PIX       (HSZ * WSZ)            // 4096

// workspace layout (bytes)
#define INT_BYTES   (BC * PIX * IN_ATOMS * 2)          // 16,777,216  bf16 input
#define WT_OFF      INT_BYTES
#define WT_BYTES    (KPOS * OC * IN_ATOMS * 2)         // 409,600     bf16 weights
#define VOTES_OFF   (WT_OFF + WT_BYTES)                // 17,186,816  fp32 votes

__device__ __forceinline__ unsigned short f2bf(float f) {
    unsigned int u = __float_as_uint(f);
    unsigned int r = u + 0x7FFFu + ((u >> 16) & 1u);   // round-to-nearest-even
    return (unsigned short)(r >> 16);
}

// input f32 [bc][ic][y][x]  ->  bf16 [bc][y][x][ic]
__global__ __launch_bounds__(256) void prep_input(const float* __restrict__ in,
                                                  unsigned short* __restrict__ outT) {
    const int t   = blockIdx.x * 256 + threadIdx.x;    // 0 .. BC*PIX-1
    const int pix = t & (PIX - 1);
    const int bc  = t >> 12;
    unsigned short tmp[IN_ATOMS];
#pragma unroll
    for (int ic = 0; ic < IN_ATOMS; ++ic)
        tmp[ic] = f2bf(in[(bc * IN_ATOMS + ic) * PIX + pix]);
#pragma unroll
    for (int ic = 0; ic < IN_ATOMS; ++ic)
        outT[t * IN_ATOMS + ic] = tmp[ic];
}

// W f32 [oc][ic][kh][kw]  ->  bf16 [kk][oc][ic]
__global__ __launch_bounds__(256) void prep_weights(const float* __restrict__ W,
                                                    unsigned short* __restrict__ Wt) {
    const int t  = blockIdx.x * 256 + threadIdx.x;     // 0 .. 204799
    const int ic = t & 31;
    const int oc = (t >> 5) & 255;
    const int kk = t >> 13;
    Wt[t] = f2bf(W[(oc * IN_ATOMS + ic) * KPOS + kk]);
}

// Implicit-GEMM conv: one wave = 4 x 16(oc) x 16(pixel) tiles, K = 800 = 25x32.
// B fragment (pixels) is reused across the 4 oc tiles -> 4 WMMAs per B load.
__global__ __launch_bounds__(256) void conv_wmma(const __bf16* __restrict__ inT,
                                                 const __bf16* __restrict__ Wt,
                                                 float* __restrict__ votes) {
    const int lane = threadIdx.x & 31;
    int w = blockIdx.x * 8 + (threadIdx.x >> 5);       // global wave id (0..65535)
    const int ocg = w & 3;   w >>= 2;                  // 4 groups of 4 oc tiles
    const int xt  = w & 3;   w >>= 2;                  // 4 x-tiles
    const int y   = w & 63;  w >>= 6;                  // 64 rows
    const int bc  = w;                                 // 64 batch-caps

    const int ln     = lane & 15;
    const int khalf  = lane >> 4;
    const int ocbase = ocg << 6;                       // 64 oc per wave
    const int xbase  = xt << 4;

    v8f acc[4];
#pragma unroll
    for (int g = 0; g < 4; ++g) { v8f z = {}; acc[g] = z; }

#pragma unroll
    for (int kk = 0; kk < KPOS; ++kk) {
        const int kh = kk / 5, kw = kk % 5;

        // B fragment: 32 ic x 16 pixels. Lane(col=ln, khalf) holds
        // K = 8*khalf..+7 and K = 16+8*khalf..+7 (channels-last input).
        const int yy = y + kh - 2;
        const int xx = xbase + ln + kw - 2;
        v16bf b = {};
        if ((unsigned)yy < (unsigned)HSZ && (unsigned)xx < (unsigned)WSZ) {
            const __bf16* ip = inT + ((((bc << 6) + yy) << 6) + xx) * IN_ATOMS
                                   + (khalf << 3);
            v8bf blo = *(const v8bf*)(ip);
            v8bf bhi = *(const v8bf*)(ip + 16);
            b = __builtin_shufflevector(blo, bhi,
                 0,1,2,3,4,5,6,7,8,9,10,11,12,13,14,15);
        }

        if (kk < KPOS - 1)   // prefetch next K-step's weight block (L2-resident)
            __builtin_prefetch(Wt + (((kk + 1) * OC + ocbase + ln) << 5), 0, 0);

        // 4 A fragments (16 oc rows x 32 ic each) against the shared B.
        // Lane(row=ln, khalf) holds K = 8*khalf..+7 and K = 16+8*khalf..+7.
#pragma unroll
        for (int g = 0; g < 4; ++g) {
            const __bf16* wp = Wt + ((kk * OC + ocbase + (g << 4) + ln) << 5)
                                  + (khalf << 3);
            v8bf alo = *(const v8bf*)(wp);
            v8bf ahi = *(const v8bf*)(wp + 16);
            v16bf a = __builtin_shufflevector(alo, ahi,
                       0,1,2,3,4,5,6,7,8,9,10,11,12,13,14,15);
            acc[g] = __builtin_amdgcn_wmma_f32_16x16x32_bf16(
                        false, a, false, b, (short)0, acc[g], false, false);
        }
    }

    // C layout: VGPR r, lane -> (m = r + 8*khalf, n = ln).
    // oc = ocbase + g*16 + 8*khalf + r  => 8 contiguous floats per group/lane.
    float* vp = votes + ((((bc << 6) + y) << 6) + xbase + ln) * OC
                      + ocbase + (khalf << 3);
#pragma unroll
    for (int g = 0; g < 4; ++g)
#pragma unroll
        for (int r = 0; r < 8; ++r) vp[(g << 4) + r] = acc[g][r];
}

// Dynamic routing: one wave per (b, y, x); lane = atom index.
// Votes staged global->LDS with CDNA5 async-to-LDS (ASYNCcnt-tracked).
__global__ __launch_bounds__(128) void routing_kernel(const float* __restrict__ votes,
                                                      const float* __restrict__ bias,
                                                      float* __restrict__ out) {
    __shared__ float Vs[4][IN_CAPS * NUM_CAPS * NUM_ATOMS]; // 4 x 2048 floats
    __shared__ float LG[4][IN_CAPS * NUM_CAPS];             // logits
    __shared__ float RT[4][IN_CAPS * NUM_CAPS];             // route

    const int lane = threadIdx.x & 31;
    const int wv   = threadIdx.x >> 5;
    const int pos  = blockIdx.x * 4 + wv;                   // 0 .. 32767
    const int b    = pos >> 12;
    const int y    = (pos >> 6) & 63;
    const int x    = pos & 63;

    float* V = Vs[wv];

    // stage votes [ic][nc][na] into LDS: 16B async copies per lane
#pragma unroll
    for (int ic = 0; ic < IN_CAPS; ++ic) {
        const float* g = votes + ((((b * IN_CAPS + ic) << 6) + y) * 64 + x) * OC;
#pragma unroll
        for (int h = 0; h < 2; ++h) {
            const int c4 = h * 32 + lane;                   // 16B chunk id (0..63)
            unsigned long long ga = (unsigned long long)(uintptr_t)(g + c4 * 4);
            unsigned lo = (unsigned)(uintptr_t)(&V[ic * OC + c4 * 4]);
            asm volatile("global_load_async_to_lds_b128 %0, %1, off"
                         :: "v"(lo), "v"(ga) : "memory");
        }
    }
    LG[wv][lane]      = 0.0f;
    LG[wv][lane + 32] = 0.0f;
    asm volatile("s_wait_asynccnt 0x0" ::: "memory");
    __syncthreads();

    float act[NUM_CAPS];
#pragma unroll
    for (int it = 0; it < 3; ++it) {
        // route = softmax over nc of logits[ic][:]
#pragma unroll
        for (int h = 0; h < 2; ++h) {
            const int p  = lane + h * 32;
            const int ic = p >> 3;
            float mx = LG[wv][ic * 8];
#pragma unroll
            for (int j = 1; j < 8; ++j) mx = fmaxf(mx, LG[wv][ic * 8 + j]);
            float s = 0.0f;
#pragma unroll
            for (int j = 0; j < 8; ++j) s += __expf(LG[wv][ic * 8 + j] - mx);
            RT[wv][p] = __expf(LG[wv][p] - mx) / s;
        }
        __syncthreads();

        // preactivate + squash (norm over atoms == cross-lane reduce)
#pragma unroll
        for (int nc = 0; nc < NUM_CAPS; ++nc) {
            float s = bias[nc * NUM_ATOMS + lane];
#pragma unroll
            for (int ic = 0; ic < IN_CAPS; ++ic)
                s += RT[wv][ic * 8 + nc] * V[(ic * 8 + nc) * 32 + lane];
            float ns = s * s;
#pragma unroll
            for (int m = 16; m >= 1; m >>= 1) ns += __shfl_xor(ns, m, 32);
            act[nc] = s * sqrtf(ns) / (1.0f + ns);   // x/|x| * |x|^2/(1+|x|^2)
        }

        if (it < 2) {
            // agreement: logits[ic][nc] += sum_na votes * activation
#pragma unroll
            for (int ic = 0; ic < IN_CAPS; ++ic) {
#pragma unroll
                for (int nc = 0; nc < NUM_CAPS; ++nc) {
                    float d = V[(ic * 8 + nc) * 32 + lane] * act[nc];
#pragma unroll
                    for (int m = 16; m >= 1; m >>= 1) d += __shfl_xor(d, m, 32);
                    if (lane == 0) LG[wv][ic * 8 + nc] += d;
                }
            }
            __syncthreads();
        }
    }

    // out [B][NC][NA][H][W]
#pragma unroll
    for (int nc = 0; nc < NUM_CAPS; ++nc)
        out[((((b * NUM_CAPS + nc) * NUM_ATOMS + lane) << 6) + y) * 64 + x] = act[nc];
}

extern "C" void kernel_launch(void* const* d_in, const int* in_sizes, int n_in,
                              void* d_out, int out_size, void* d_ws, size_t ws_size,
                              hipStream_t stream) {
    const float* input = (const float*)d_in[0];   // [8,8,32,64,64]
    const float* W     = (const float*)d_in[1];   // [256,32,5,5]
    const float* bias  = (const float*)d_in[2];   // [1,1,8,32]
    float*       out   = (float*)d_out;           // [8,8,32,64,64]

    char*   ws    = (char*)d_ws;
    __bf16* inT   = (__bf16*)(ws);
    __bf16* Wt    = (__bf16*)(ws + WT_OFF);
    float*  votes = (float*)(ws + VOTES_OFF);

    // 64*4096 = 262144 threads
    prep_input  <<<1024, 256, 0, stream>>>(input, (unsigned short*)inT);
    // 25*256*32 = 204800 threads
    prep_weights<<<800, 256, 0, stream>>>(W, (unsigned short*)Wt);
    // 65536 wave-tiles (4 oc-tiles each) / 8 waves per block
    conv_wmma   <<<8192, 256, 0, stream>>>(inT, Wt, votes);
    // 32768 positions / 4 waves per block
    routing_kernel<<<8192, 128, 0, stream>>>(votes, bias, out);
}
